// CrossViewCrossAttention_38268158607378
// MI455X (gfx1250) — matrix operand order repaired and checked by reading, not compiled, not run on hardware
//
#include <hip/hip_runtime.h>
#include <hip/hip_bf16.h>
#include <math.h>

#define DEV __device__ __forceinline__

typedef __attribute__((ext_vector_type(16))) _Float16 v16h;
typedef __attribute__((ext_vector_type(8)))  _Float16 v8h;
typedef __attribute__((ext_vector_type(8)))  float    v8f;

static constexpr int Bn = 8, Cc = 256, NX = 1024, NZ = 256, Hn = 8, Dh = 32, MIDn = 1024;
static constexpr int MX = Bn * NX;   // 8192 query rows
static constexpr int MZ = Bn * NZ;   // 2048 uav rows

// ---------------------------------------------------------------------------
// WMMA fragment loaders (CDNA5 v_wmma_f32_16x16x32_f16 layouts, wave32)
// A (16x32, MxK): lane<16: row=lane, K={k0..k0+7, k0+16..k0+23}
//                 lane>=16: row=lane-16, K={k0+8..k0+15, k0+24..k0+31}
// B (32x16, KxN): lane<16: col=lane, K={k0..+7, k0+16..+23} ; lane>=16: col=lane-16, K={k0+8..+15, k0+24..+31}
// C/D (16x16 f32): lane<16: N=lane, rows 0..7 in v0..7 ; lane>=16: N=lane-16, rows 8..15
// ---------------------------------------------------------------------------
DEV v16h load_fragA(const _Float16* base, int row0, int stride, int k0) {
  const int lane = threadIdx.x & 31;
  const int half = lane >> 4;
  const int r = lane & 15;
  const _Float16* p = base + (long)(row0 + r) * stride + k0 + half * 8;
  v8h lo = *(const v8h*)p;         // K = kb .. kb+7
  v8h hi = *(const v8h*)(p + 16);  // K = kb+16 .. kb+23
  v16h out;
#pragma unroll
  for (int i = 0; i < 8; ++i) { out[i] = lo[i]; out[i + 8] = hi[i]; }
  return out;
}

DEV v16h load_fragB(const _Float16* base, int col0, int stride, int k0) {
  const int lane = threadIdx.x & 31;
  const int half = lane >> 4;
  const int c = lane & 15;
  const _Float16* p = base + (long)(col0 + c) * stride + k0 + half * 8;
  v8h lo = *(const v8h*)p;         // K = k0+half*8 .. +7
  v8h hi = *(const v8h*)(p + 16);  // K = k0+16+half*8 .. +7
  v16h out;
#pragma unroll
  for (int i = 0; i < 8; ++i) { out[i] = lo[i]; out[i + 8] = hi[i]; }
  return out;
}

DEV v8f wmma16(v16h a, v16h b, v8f c) {
  return __builtin_amdgcn_wmma_f32_16x16x32_f16(false, a, false, b, (short)0, c, false, false);
}

// ---------------------------------------------------------------------------
// Prep kernels
// ---------------------------------------------------------------------------
__global__ void k_convert_f16(const float* __restrict__ src, _Float16* __restrict__ dst, int n) {
  int i = blockIdx.x * blockDim.x + threadIdx.x;
  if (i < n) dst[i] = (_Float16)src[i];
}

// (B, C, N) f32 -> (B, N, C) f16
__global__ void k_pack_nc(const float* __restrict__ src, _Float16* __restrict__ dst,
                          int Bv, int Cv, int Nv) {
  int t = blockIdx.x * blockDim.x + threadIdx.x;
  if (t >= Bv * Cv * Nv) return;
  int n = t % Nv;
  int c = (t / Nv) % Cv;
  int b = t / (Nv * Cv);
  dst[((long)b * Nv + n) * Cv + c] = (_Float16)src[t];
}

// ---------------------------------------------------------------------------
// Register-blocked WMMA GEMM: 64(M) x 64(N) per wave, 16 accumulators.
// B fragments loaded once per K-step and reused by 4 A tiles.
// Y[m,n] = act( (sum_k X[m,k]*W[n,k] + bias[n]) * out_scale )
// out_mode 0: f16 row-major [M,N]
// out_mode 1: f16 V-transposed store: m=(b,key), Y[(b*N+n)*Lk + key]  (per-head V^T)
// out_mode 2: f32 row-major [M,N]
// act 1: exact GELU
// Requires M % 64 == 0, N % 64 == 0, K % 32 == 0.
// ---------------------------------------------------------------------------
__global__ void __launch_bounds__(128)
k_gemm(const _Float16* __restrict__ X, const _Float16* __restrict__ W,
       const float* __restrict__ bias, void* __restrict__ Yp,
       int M, int N, int K, float out_scale, int out_mode, int act, int Lkv) {
  int wave = (blockIdx.x * blockDim.x + threadIdx.x) >> 5;
  int tilesN = N >> 6;
  int tm = wave / tilesN;
  int tn = wave - tm * tilesN;
  if (tm >= (M >> 6)) return;
  int m0 = tm << 6, n0 = tn << 6;

  v8f acc[4][4] = {};
  for (int k0 = 0; k0 < K; k0 += 32) {
    v16h b0 = load_fragB(W, n0, K, k0);
    v16h b1 = load_fragB(W, n0 + 16, K, k0);
    v16h b2 = load_fragB(W, n0 + 32, K, k0);
    v16h b3 = load_fragB(W, n0 + 48, K, k0);
#pragma unroll
    for (int i = 0; i < 4; ++i) {
      v16h a = load_fragA(X, m0 + i * 16, K, k0);
      acc[i][0] = wmma16(a, b0, acc[i][0]);
      acc[i][1] = wmma16(a, b1, acc[i][1]);
      acc[i][2] = wmma16(a, b2, acc[i][2]);
      acc[i][3] = wmma16(a, b3, acc[i][3]);
    }
  }

  int lane = threadIdx.x & 31;
  int half = lane >> 4, c = lane & 15;
#pragma unroll
  for (int j = 0; j < 4; ++j) {
    int n = n0 + j * 16 + c;
    float bv = bias[n];
#pragma unroll
    for (int i = 0; i < 4; ++i) {
#pragma unroll
      for (int r = 0; r < 8; ++r) {
        int m = m0 + i * 16 + half * 8 + r;
        float v = (acc[i][j][r] + bv) * out_scale;
        if (act == 1) v = 0.5f * v * (1.0f + erff(v * 0.70710678118654752f));
        if (out_mode == 0) {
          ((_Float16*)Yp)[(long)m * N + n] = (_Float16)v;
        } else if (out_mode == 1) {
          int bb = m / Lkv, key = m - bb * Lkv;
          ((_Float16*)Yp)[((long)bb * N + n) * Lkv + key] = (_Float16)v;
        } else {
          ((float*)Yp)[(long)m * N + n] = v;
        }
      }
    }
  }
}

// ---------------------------------------------------------------------------
// Flash attention: one wave per (b, h, 32-query tile); key blocks of 32.
// Two 16-row Q tiles share every K/V fragment (8 WMMA per 8 b128 loads).
// qp (B,Lq,C) f16 already scaled by 1/sqrt(d); kp (B,Lk,C) f16; vT (B,H,Dh,Lk) f16
// oh (B,Lq,C) f16 = softmax(QK^T) V, heads concatenated.
// ml (optional): [B*H*Lq] row max then [B*H*Lq] row sum-of-exp (for coarse_cls).
// ---------------------------------------------------------------------------
__global__ void __launch_bounds__(128)
k_flash(const _Float16* __restrict__ qp, const _Float16* __restrict__ kp,
        const _Float16* __restrict__ vT, _Float16* __restrict__ oh,
        float* __restrict__ ml, int Lq, int Lk) {
  __shared__ __align__(16) _Float16 Pb[4][32][40];
  const int wl = threadIdx.x >> 5;
  const int lane = threadIdx.x & 31;
  const int half = lane >> 4, c = lane & 15;
  int gwave = blockIdx.x * 4 + wl;
  int qtiles = Lq >> 5;
  int qt = gwave % qtiles;
  int h = (gwave / qtiles) % Hn;
  int b = gwave / (qtiles * Hn);

  const _Float16* qbase = qp + (long)b * Lq * Cc;
  v16h qf0 = load_fragA(qbase, qt * 32, Cc, h * Dh);
  v16h qf1 = load_fragA(qbase, qt * 32 + 16, Cc, h * Dh);
  const _Float16* kbase = kp + (long)b * Lk * Cc;
  const _Float16* vbase = vT + ((long)b * Cc + h * Dh) * Lk;

  float m_run[2][8], l_run[2][8];
  v8f acc[2][2] = {};
#pragma unroll
  for (int i = 0; i < 2; ++i)
#pragma unroll
    for (int r = 0; r < 8; ++r) { m_run[i][r] = -3.0e38f; l_run[i][r] = 0.0f; }

  for (int kt = 0; kt < Lk; kt += 32) {
    if (kt + 32 < Lk) {  // global_prefetch_b8 for next key block (K rows + V^T rows)
      __builtin_prefetch(kbase + (long)(kt + 32 + lane) * Cc + h * Dh, 0, 3);
      __builtin_prefetch(vbase + (long)lane * Lk + kt + 32, 0, 3);
    }
    v16h bk0 = load_fragB(kbase, kt, Cc, h * Dh);
    v16h bk1 = load_fragB(kbase, kt + 16, Cc, h * Dh);
    v8f zero = {};
    v8f s[2][2];
    s[0][0] = wmma16(qf0, bk0, zero);
    s[0][1] = wmma16(qf0, bk1, zero);
    s[1][0] = wmma16(qf1, bk0, zero);
    s[1][1] = wmma16(qf1, bk1, zero);
    // online softmax; each half-wave owns rows half*8 + r, 16 lanes span keys
#pragma unroll
    for (int i = 0; i < 2; ++i) {
#pragma unroll
      for (int r = 0; r < 8; ++r) {
        float mx = fmaxf(s[i][0][r], s[i][1][r]);
        mx = fmaxf(mx, __shfl_xor(mx, 1));
        mx = fmaxf(mx, __shfl_xor(mx, 2));
        mx = fmaxf(mx, __shfl_xor(mx, 4));
        mx = fmaxf(mx, __shfl_xor(mx, 8));
        float nm = fmaxf(m_run[i][r], mx);
        float sf = __expf(m_run[i][r] - nm);
        m_run[i][r] = nm;
        float p0 = __expf(s[i][0][r] - nm);
        float p1 = __expf(s[i][1][r] - nm);
        s[i][0][r] = p0; s[i][1][r] = p1;
        float ls = p0 + p1;
        ls += __shfl_xor(ls, 1);
        ls += __shfl_xor(ls, 2);
        ls += __shfl_xor(ls, 4);
        ls += __shfl_xor(ls, 8);
        l_run[i][r] = l_run[i][r] * sf + ls;
        acc[i][0][r] *= sf;
        acc[i][1][r] *= sf;
      }
    }
    // transpose P (32q x 32k) through LDS into A-fragment layout
    __syncthreads();
#pragma unroll
    for (int i = 0; i < 2; ++i) {
#pragma unroll
      for (int r = 0; r < 8; ++r) {
        Pb[wl][i * 16 + half * 8 + r][c] = (_Float16)s[i][0][r];
        Pb[wl][i * 16 + half * 8 + r][16 + c] = (_Float16)s[i][1][r];
      }
    }
    __syncthreads();
    v16h pf0, pf1;
    {
      const _Float16* pp0 = &Pb[wl][c][half * 8];
      const _Float16* pp1 = &Pb[wl][16 + c][half * 8];
      v8h lo0 = *(const v8h*)pp0;
      v8h hi0 = *(const v8h*)(pp0 + 16);
      v8h lo1 = *(const v8h*)pp1;
      v8h hi1 = *(const v8h*)(pp1 + 16);
#pragma unroll
      for (int i = 0; i < 8; ++i) {
        pf0[i] = lo0[i]; pf0[i + 8] = hi0[i];
        pf1[i] = lo1[i]; pf1[i + 8] = hi1[i];
      }
    }
    v16h v0 = load_fragB(vbase, 0, Lk, kt);   // out dims 0..15
    v16h v1 = load_fragB(vbase, 16, Lk, kt);  // out dims 16..31
    acc[0][0] = wmma16(pf0, v0, acc[0][0]);
    acc[0][1] = wmma16(pf0, v1, acc[0][1]);
    acc[1][0] = wmma16(pf1, v0, acc[1][0]);
    acc[1][1] = wmma16(pf1, v1, acc[1][1]);
  }

#pragma unroll
  for (int i = 0; i < 2; ++i) {
#pragma unroll
    for (int r = 0; r < 8; ++r) {
      float inv = 1.0f / l_run[i][r];
      int q = qt * 32 + i * 16 + half * 8 + r;
      long orow = ((long)b * Lq + q) * Cc + h * Dh;
      oh[orow + c] = (_Float16)(acc[i][0][r] * inv);
      oh[orow + 16 + c] = (_Float16)(acc[i][1][r] * inv);
    }
  }
  if (ml != nullptr && c == 0) {
#pragma unroll
    for (int i = 0; i < 2; ++i) {
#pragma unroll
      for (int r = 0; r < 8; ++r) {
        int q = qt * 32 + i * 16 + half * 8 + r;
        ml[((long)b * Hn + h) * Lq + q] = m_run[i][r];
        ml[(long)Bn * Hn * Lq + ((long)b * Hn + h) * Lq + q] = l_run[i][r];
      }
    }
  }
}

// ---------------------------------------------------------------------------
// coarse_cls[b,q] = mean_h softmax(scores)[q, center]; recompute single column
// from scaled qp, kp and saved (m, l).
// ---------------------------------------------------------------------------
__global__ void k_coarse(const _Float16* __restrict__ qp, const _Float16* __restrict__ kp,
                         const float* __restrict__ ml, float* __restrict__ outp) {
  int t = blockIdx.x * blockDim.x + threadIdx.x;
  if (t >= Bn * NX) return;
  int b = t >> 10, q = t & 1023;
  const _Float16* qr = qp + ((long)b * NX + q) * Cc;
  const _Float16* kr = kp + ((long)b * NZ + (NZ / 2)) * Cc;
  float accum = 0.0f;
  for (int h = 0; h < Hn; ++h) {
    float s = 0.0f;
#pragma unroll
    for (int dd = 0; dd < Dh; ++dd)
      s += (float)qr[h * Dh + dd] * (float)kr[h * Dh + dd];
    float m = ml[((long)b * Hn + h) * NX + q];
    float l = ml[(long)Bn * Hn * NX + ((long)b * Hn + h) * NX + q];
    accum += __expf(s - m) / l;
  }
  outp[t] = accum * (1.0f / Hn);
}

// ---------------------------------------------------------------------------
// LayerNorm over C=256, one wave per row; f32 in -> f16 out.
// ---------------------------------------------------------------------------
__global__ void __launch_bounds__(128)
k_layernorm(const float* __restrict__ xin, const float* __restrict__ g,
            const float* __restrict__ bta, _Float16* __restrict__ yout, int Mrows) {
  int row = blockIdx.x * 4 + (threadIdx.x >> 5);
  int lane = threadIdx.x & 31;
  if (row >= Mrows) return;
  const float* xr = xin + (long)row * Cc;
  float vals[8];
  float s = 0.0f;
#pragma unroll
  for (int i = 0; i < 8; ++i) { vals[i] = xr[lane * 8 + i]; s += vals[i]; }
  s += __shfl_xor(s, 16); s += __shfl_xor(s, 8); s += __shfl_xor(s, 4);
  s += __shfl_xor(s, 2);  s += __shfl_xor(s, 1);
  float mean = s * (1.0f / Cc);
  float vs = 0.0f;
#pragma unroll
  for (int i = 0; i < 8; ++i) { float d = vals[i] - mean; vs += d * d; }
  vs += __shfl_xor(vs, 16); vs += __shfl_xor(vs, 8); vs += __shfl_xor(vs, 4);
  vs += __shfl_xor(vs, 2);  vs += __shfl_xor(vs, 1);
  float inv = rsqrtf(vs * (1.0f / Cc) + 1e-5f);
#pragma unroll
  for (int i = 0; i < 8; ++i) {
    int col = lane * 8 + i;
    yout[(long)row * Cc + col] = (_Float16)((vals[i] - mean) * inv * g[col] + bta[col]);
  }
}

// ---------------------------------------------------------------------------
// Final 2-wide regression: loc_bias[b, j, q] = reg_b[j] + fine[b,q,:] . reg_w[j,:]
// ---------------------------------------------------------------------------
__global__ void k_reg(const _Float16* __restrict__ fine, const float* __restrict__ rw,
                      const float* __restrict__ rb, float* __restrict__ outp) {
  int t = blockIdx.x * blockDim.x + threadIdx.x;
  if (t >= Bn * 2 * NX) return;
  int b = t >> 11;
  int j = (t >> 10) & 1;
  int q = t & 1023;
  const _Float16* fr = fine + ((long)b * NX + q) * MIDn;
  const float* wr = rw + j * MIDn;
  float accum = rb[j];
  for (int k = 0; k < MIDn; ++k) accum += (float)fr[k] * wr[k];
  outp[t] = accum;  // layout (B, 2, 32, 32) flattened == b*2048 + j*1024 + q
}

// ---------------------------------------------------------------------------
extern "C" void kernel_launch(void* const* d_in, const int* in_sizes, int n_in,
                              void* d_out, int out_size, void* d_ws, size_t ws_size,
                              hipStream_t stream) {
  const float* z        = (const float*)d_in[0];
  const float* x        = (const float*)d_in[1];
  const float* ca_w_in  = (const float*)d_in[2];
  const float* ca_b_in  = (const float*)d_in[3];
  const float* ca_w_out = (const float*)d_in[4];
  const float* ca_b_out = (const float*)d_in[5];
  const float* ca_ln_g  = (const float*)d_in[6];
  const float* ca_ln_b  = (const float*)d_in[7];
  const float* sa_w_in  = (const float*)d_in[8];
  const float* sa_b_in  = (const float*)d_in[9];
  const float* sa_w_out = (const float*)d_in[10];
  const float* sa_b_out = (const float*)d_in[11];
  const float* ln_g     = (const float*)d_in[12];
  const float* ln_b     = (const float*)d_in[13];
  const float* fine_w   = (const float*)d_in[14];
  const float* fine_b   = (const float*)d_in[15];
  const float* reg_w    = (const float*)d_in[16];
  const float* reg_b    = (const float*)d_in[17];
  float* outp = (float*)d_out;
  char* ws = (char*)d_ws;

  // workspace layout (bytes)
  _Float16* wcain  = (_Float16*)(ws + 0);          // 768*256        -> 393216
  _Float16* wcaout = (_Float16*)(ws + 393216);     // 256*256        -> 524288
  _Float16* wsain  = (_Float16*)(ws + 524288);     // 2*768*256      -> 1310720
  _Float16* wsaout = (_Float16*)(ws + 1310720);    // 2*256*256      -> 1572864
  _Float16* wfine  = (_Float16*)(ws + 1572864);    // 1024*256       -> 2097152
  _Float16* xh     = (_Float16*)(ws + 2097152);    // 8192*256       -> 6291456
  _Float16* zh     = (_Float16*)(ws + 6291456);    // 2048*256       -> 7340032
  _Float16* qp     = (_Float16*)(ws + 7340032);    // 8192*256       -> 11534336
  _Float16* kp     = (_Float16*)(ws + 11534336);   // 8192*256       -> 15728640
  _Float16* vT     = (_Float16*)(ws + 15728640);   // 8*256*1024     -> 19922944
  _Float16* oh     = (_Float16*)(ws + 19922944);   // 8192*256       -> 24117248
  float*    y32    = (float*)   (ws + 24117248);   // 8192*256 f32   -> 32505856
  _Float16* yh     = (_Float16*)(ws + 32505856);   // 8192*256       -> 36700160
  float*    ml     = (float*)   (ws + 36700160);   // 2*8*8*1024 f32 -> 37224448
  _Float16* fineb  = (_Float16*)(ws + 37224448);   // 8192*1024      -> 54001664

  const float qscale = 0.1767766952966369f;  // 1/sqrt(32)
  auto cdiv = [](int a, int b) { return (a + b - 1) / b; };

  // --- weight conversion + input packing ---
  k_convert_f16<<<cdiv(768 * 256, 256), 256, 0, stream>>>(ca_w_in, wcain, 768 * 256);
  k_convert_f16<<<cdiv(256 * 256, 256), 256, 0, stream>>>(ca_w_out, wcaout, 256 * 256);
  k_convert_f16<<<cdiv(2 * 768 * 256, 256), 256, 0, stream>>>(sa_w_in, wsain, 2 * 768 * 256);
  k_convert_f16<<<cdiv(2 * 256 * 256, 256), 256, 0, stream>>>(sa_w_out, wsaout, 2 * 256 * 256);
  k_convert_f16<<<cdiv(1024 * 256, 256), 256, 0, stream>>>(fine_w, wfine, 1024 * 256);
  k_pack_nc<<<cdiv(Bn * Cc * NX, 256), 256, 0, stream>>>(x, xh, Bn, Cc, NX);
  k_pack_nc<<<cdiv(Bn * Cc * NZ, 256), 256, 0, stream>>>(z, zh, Bn, Cc, NZ);

  auto gemm = [&](const _Float16* X, const _Float16* W, const float* bias, void* Y,
                  int M, int N, int K, float sc, int mode, int act, int Lkv) {
    int waves = (M >> 6) * (N >> 6);   // 64x64 per wave
    k_gemm<<<waves / 4, 128, 0, stream>>>(X, W, bias, Y, M, N, K, sc, mode, act, Lkv);
  };
  auto flash = [&](const _Float16* Q, const _Float16* Kf, const _Float16* Vf,
                   _Float16* O, float* mlv, int Lq, int Lk) {
    int waves = Bn * Hn * (Lq >> 5);   // 32 queries per wave
    k_flash<<<waves / 4, 128, 0, stream>>>(Q, Kf, Vf, O, mlv, Lq, Lk);
  };

  // --- cross attention: Q from x, K/V from z ---
  gemm(xh, wcain,              ca_b_in,       qp, MX, Cc, Cc, qscale, 0, 0, 0);
  gemm(zh, wcain + 256 * 256,  ca_b_in + 256, kp, MZ, Cc, Cc, 1.0f,   0, 0, 0);
  gemm(zh, wcain + 512 * 256,  ca_b_in + 512, vT, MZ, Cc, Cc, 1.0f,   1, 0, NZ);
  flash(qp, kp, vT, oh, ml, NX, NZ);
  gemm(oh, wcaout, ca_b_out, y32, MX, Cc, Cc, 1.0f, 2, 0, 0);
  k_coarse<<<cdiv(Bn * NX, 256), 256, 0, stream>>>(qp, kp, ml, outp);
  k_layernorm<<<MX / 4, 128, 0, stream>>>(y32, ca_ln_g, ca_ln_b, yh, MX);

  // --- 2 self-attention layers ---
  for (int i = 0; i < 2; ++i) {
    const _Float16* wi = wsain + (long)i * 768 * 256;
    const float* bi = sa_b_in + i * 768;
    gemm(yh, wi,             bi,       qp, MX, Cc, Cc, qscale, 0, 0, 0);
    gemm(yh, wi + 256 * 256, bi + 256, kp, MX, Cc, Cc, 1.0f,   0, 0, 0);
    gemm(yh, wi + 512 * 256, bi + 512, vT, MX, Cc, Cc, 1.0f,   1, 0, NX);
    flash(qp, kp, vT, oh, nullptr, NX, NX);
    gemm(oh, wsaout + (long)i * 256 * 256, sa_b_out + i * 256, y32, MX, Cc, Cc, 1.0f, 2, 0, 0);
    k_layernorm<<<MX / 4, 128, 0, stream>>>(y32, ln_g + i * 256, ln_b + i * 256, yh, MX);
  }

  // --- fine MLP (GELU) + 2-wide regression ---
  gemm(yh, wfine, fine_b, fineb, MX, MIDn, Cc, 1.0f, 0, 1, 0);
  k_reg<<<cdiv(Bn * 2 * NX, 256), 256, 0, stream>>>(fineb, reg_w, reg_b, outp + Bn * NX);
}
